// T5EncoderSelfAttention_79688823210503
// MI455X (gfx1250) — matrix operand order repaired and checked by reading, not compiled, not run on hardware
//
#include <hip/hip_runtime.h>
#include <hip/hip_bf16.h>

typedef __bf16 bf16_t;
typedef __bf16 bf16x8  __attribute__((ext_vector_type(8)));
typedef __bf16 bf16x16 __attribute__((ext_vector_type(16)));
typedef float  floatx4 __attribute__((ext_vector_type(4)));
typedef float  floatx8 __attribute__((ext_vector_type(8)));
typedef unsigned int uint32x4 __attribute__((ext_vector_type(4)));
typedef int          int32x4  __attribute__((ext_vector_type(4)));
typedef int          int32x8  __attribute__((ext_vector_type(8)));

#define NB    2
#define NS    2048
#define NHID  1024
#define NH    16
#define ND    64

#if defined(__has_builtin)
#  if __has_builtin(__builtin_amdgcn_tensor_load_to_lds) && \
      __has_builtin(__builtin_amdgcn_s_wait_tensorcnt)
#    define HAVE_TDM 1
#  endif
#endif
#ifndef HAVE_TDM
#  define HAVE_TDM 0
#endif

static __device__ inline unsigned short f2bf_bits(float f) {
  bf16_t h = (bf16_t)f;
  return __builtin_bit_cast(unsigned short, h);
}
static __device__ inline unsigned short bf_bits(bf16_t h) {
  return __builtin_bit_cast(unsigned short, h);
}
static __device__ inline bf16x16 frag2(const bf16_t* p0, const bf16_t* p1) {
  bf16x8 lo = *(const bf16x8*)p0;
  bf16x8 hi = *(const bf16x8*)p1;
  return __builtin_shufflevector(lo, hi, 0,1,2,3,4,5,6,7,8,9,10,11,12,13,14,15);
}
static __device__ inline floatx8 wmma_bf16(bf16x16 a, bf16x16 b, floatx8 c) {
  return __builtin_amdgcn_wmma_f32_16x16x32_bf16(false, a, false, b, (short)0, c,
                                                 false, false);
}
static __device__ inline floatx8 zero8() {
  floatx8 z = {0.f,0.f,0.f,0.f,0.f,0.f,0.f,0.f};
  return z;
}

// ---------------------------------------------------------------------------
// NT GEMM: C[M,N] = A[M,K] * B[N,K]^T, bf16 WMMA, f32 accumulate.
// MODE 0: A is f32 (x), epilogue scatters bf16 into qkv [3,B,H,S,D] layout.
// MODE 1: A is bf16 (attention out), epilogue stores f32 row-major [M,N].
// Block tile 128x128, BK=32, 8 waves in 2x4 -> each wave 64x32 (4x2 WMMA tiles)
// ---------------------------------------------------------------------------
template<int MODE>
__global__ __launch_bounds__(256) void gemm_nt(
    const void* __restrict__ Ain, const float* __restrict__ Bw,
    void* __restrict__ Cout, int M, int N, int K)
{
  __shared__ bf16_t lA[128 * 48];   // [m][k], row stride 48 bf16 (96 B)
  __shared__ bf16_t lB[32 * 136];   // [k][n], row stride 136 bf16 (272 B)
  unsigned* lBu = (unsigned*)lB;

  const int tid  = threadIdx.x;
  const int w    = tid >> 5, lane = tid & 31;
  const int hl   = lane >> 4, lr = lane & 15;
  const int wrow = w >> 2, wcol = w & 3;
  const int blockM = blockIdx.y * 128, blockN = blockIdx.x * 128;

  floatx8 acc[4][2];
  for (int mt = 0; mt < 4; ++mt)
    for (int nt = 0; nt < 2; ++nt) acc[mt][nt] = zero8();

  const int am = tid & 127, ak = (tid >> 7) * 16;     // A staging: 16 k/thread
  const int bn = (tid & 63) * 2, bk = (tid >> 6) * 8; // B staging: 2 n x 8 k

  for (int k0 = 0; k0 < K; k0 += 32) {
    __syncthreads();
    // ---- stage A tile (convert to bf16 if needed) ----
    if (MODE == 0) {
      const float* src = (const float*)Ain + (size_t)(blockM + am) * K + k0 + ak;
      floatx4 f0 = *(const floatx4*)(src);
      floatx4 f1 = *(const floatx4*)(src + 4);
      floatx4 f2 = *(const floatx4*)(src + 8);
      floatx4 f3 = *(const floatx4*)(src + 12);
      bf16x8 v0, v1;
      for (int i = 0; i < 4; ++i) {
        v0[i]   = (bf16_t)f0[i];  v0[i+4] = (bf16_t)f1[i];
        v1[i]   = (bf16_t)f2[i];  v1[i+4] = (bf16_t)f3[i];
      }
      *(bf16x8*)&lA[am*48 + ak]     = v0;
      *(bf16x8*)&lA[am*48 + ak + 8] = v1;
    } else {
      const bf16_t* src = (const bf16_t*)Ain + (size_t)(blockM + am) * K + k0 + ak;
      *(bf16x8*)&lA[am*48 + ak]     = *(const bf16x8*)(src);
      *(bf16x8*)&lA[am*48 + ak + 8] = *(const bf16x8*)(src + 8);
    }
    // ---- stage B tile transposed to [k][n], packed u32 stores ----
    {
      const float* s0 = Bw + (size_t)(blockN + bn) * K + k0 + bk;
      const float* s1 = s0 + K;
      floatx4 a0 = *(const floatx4*)s0, a1 = *(const floatx4*)(s0 + 4);
      floatx4 c0 = *(const floatx4*)s1, c1 = *(const floatx4*)(s1 + 4);
      for (int i = 0; i < 4; ++i) {
        unsigned u0 = f2bf_bits(a0[i]) | ((unsigned)f2bf_bits(c0[i]) << 16);
        unsigned u1 = f2bf_bits(a1[i]) | ((unsigned)f2bf_bits(c1[i]) << 16);
        lBu[((bk + i    ) * 136 + bn) >> 1] = u0;
        lBu[((bk + i + 4) * 136 + bn) >> 1] = u1;
      }
    }
    // prefetch next A tile (emits global_prefetch_b8)
    if (k0 + 32 < K) {
      const char* pf = (const char*)Ain +
          ((size_t)(blockM + am) * K + k0 + 32 + ak) * (MODE == 0 ? 4 : 2);
      __builtin_prefetch(pf, 0, 1);
    }
    __syncthreads();

    // ---- fragments + 8 WMMAs per wave ----
    bf16x16 afr[4], bfr[2];
    for (int mt = 0; mt < 4; ++mt) {
      int m = wrow*64 + mt*16 + lr;
      afr[mt] = frag2(&lA[m*48 + hl*8], &lA[m*48 + hl*8 + 16]);
    }
    for (int nt = 0; nt < 2; ++nt) {
      int n0 = wcol*32 + nt*16;
      for (int s = 0; s < 16; ++s)
        bfr[nt][s] = lB[(hl*8 + (s & 7) + (s >> 3)*16)*136 + n0 + lr];
    }
    for (int mt = 0; mt < 4; ++mt)
      for (int nt = 0; nt < 2; ++nt)
        acc[mt][nt] = wmma_bf16(afr[mt], bfr[nt], acc[mt][nt]);
  }

  // ---- epilogue ----
  for (int mt = 0; mt < 4; ++mt)
    for (int nt = 0; nt < 2; ++nt)
      for (int r = 0; r < 8; ++r) {
        int m_g = blockM + wrow*64 + mt*16 + hl*8 + r;
        int n_g = blockN + wcol*32 + nt*16 + lr;
        float v = acc[mt][nt][r];
        if (MODE == 0) {
          int three = n_g >> 10, rem = n_g & 1023;
          int hh = rem >> 6, dd = rem & 63;
          int bb = m_g >> 11, ss = m_g & 2047;
          ((bf16_t*)Cout)[((((size_t)three*NB + bb)*NH + hh)*NS + ss)*ND + dd] =
              (bf16_t)v;
        } else {
          ((float*)Cout)[(size_t)m_g * N + n_g] = v;
        }
      }
}

// ---------------------------------------------------------------------------
// Flash attention: one block per (b,h, 128-query tile). 8 waves x 16 q rows.
// Streams 64-key tiles; scores = Q K^T + bias(mask), online softmax, P x V.
// V tile is staged via the Tensor Data Mover (TDM) when available; its LDS
// padding feature reproduces the 72-bf16 row stride (32-DWORD interval,
// 4-DWORD pad) so the WMMA fragment reads are identical on both paths.
// ---------------------------------------------------------------------------
__global__ __launch_bounds__(256) void attn_kernel(
    const bf16_t* __restrict__ qkv, const float* __restrict__ bias,
    const unsigned char* __restrict__ mask, bf16_t* __restrict__ aout)
{
  __shared__ bf16_t lK[64 * 72];      // K^T tile: [d][key], stride 72 (144 B)
  __shared__ bf16_t lV[64 * 72];      // V tile:   [key][d], stride 72
  __shared__ bf16_t lP[8][16 * 72];   // per-wave probs (bf16), stride 72
  unsigned* lKu = (unsigned*)lK;

  const int tid  = threadIdx.x;
  const int w    = tid >> 5, lane = tid & 31;
  const int hl   = lane >> 4, lr = lane & 15;
  const int bh   = blockIdx.y, b = bh >> 4, h = bh & 15;
  const int qbase = blockIdx.x * 128;

  const size_t plane = (size_t)NS * ND;
  const bf16_t* qp = qkv + ((size_t)(0*NB + b)*NH + h) * plane;
  const bf16_t* kp = qkv + ((size_t)(1*NB + b)*NH + h) * plane;
  const bf16_t* vp = qkv + ((size_t)(2*NB + b)*NH + h) * plane;

  // resident Q fragments: 16 rows x 64 d -> two 16x32 A frags
  const int qrow = qbase + w * 16;
  bf16x16 qf[2];
  for (int f = 0; f < 2; ++f) {
    const bf16_t* p = qp + (size_t)(qrow + lr) * ND + f*32 + hl*8;
    qf[f] = frag2(p, p + 16);
  }

  float mrun[8], lrun[8];
  floatx8 o[4];
  for (int r = 0; r < 8; ++r) { mrun[r] = -3.0e38f; lrun[r] = 0.f; }
  for (int dt = 0; dt < 4; ++dt) o[dt] = zero8();

  const int kp2 = (tid & 31) * 2, kd0 = (tid >> 5) * 8;  // K^T staging
  const int vkey = tid & 63,      vd0 = (tid >> 6) * 16; // V staging (fallback)

#if HAVE_TDM
  // Low 32 bits of a flat LDS pointer are the LDS byte address (ISA 10.2).
  const unsigned lv_lds = (unsigned)(unsigned long long)(const void*)&lV[0];
#endif

  for (int kt = 0; kt < NS / 64; ++kt) {
    const int key0 = kt * 64;
    __syncthreads();   // previous tile fully consumed before restaging

#if HAVE_TDM
    // ---- V tile via Tensor Data Mover (wave 0 issues; EXEC is ignored) ----
    if (w == 0) {
      unsigned long long ga = (unsigned long long)(vp + (size_t)key0 * ND);
      uint32x4 g0 = {
        1u,                                   // count=1, user descriptor
        lv_lds,                               // lds_addr (bytes)
        (unsigned)(ga & 0xFFFFFFFFull),       // global_addr[31:0]
        (unsigned)((ga >> 32) & 0x1FFFFFFull) | (2u << 30)  // [56:32] + type=2
      };
      int32x8 g1 = {
        (int)((1u << 16) |                    // data_size = 2 bytes (bf16)
              (1u << 20) |                    // pad_enable
              (4u << 22) |                    // pad_interval = 32 DWORDs (128 B)
              (3u << 25)),                    // pad_amount  =  4 DWORDs (16 B)
        (int)((unsigned)ND << 16),            // tensor_dim0 = 64  [79:48]
        (int)((unsigned)NS << 16),            // tensor_dim1 = 2048 [111:80]
        (int)((unsigned)ND << 16),            // tile_dim0 = 64    [127:112]
        (int)ND,                              // tile_dim1 = 64    [143:128]
        (int)ND,                              // tensor_dim0_stride = 64
        0, 0
      };
      int32x4 g2 = {0,0,0,0};
      int32x4 g3 = {0,0,0,0};
#if __clang_major__ >= 23
      int32x8 g4 = {0,0,0,0,0,0,0,0};
      __builtin_amdgcn_tensor_load_to_lds(g0, g1, g2, g3, g4, 0);
#else
      __builtin_amdgcn_tensor_load_to_lds(g0, g1, g2, g3, 0);
#endif
    }
#endif

    // ---- stage K^T to [d][key] (packed pair stores) ----
    {
      const bf16_t* s0 = kp + (size_t)(key0 + kp2) * ND + kd0;
      bf16x8 ra = *(const bf16x8*)s0;
      bf16x8 rb = *(const bf16x8*)(s0 + ND);
      for (int i = 0; i < 8; ++i) {
        unsigned u = bf_bits(ra[i]) | ((unsigned)bf_bits(rb[i]) << 16);
        lKu[((kd0 + i) * 72 + kp2) >> 1] = u;
      }
    }
#if !HAVE_TDM
    // ---- stage V straight copy [key][d] (fallback path) ----
    {
      const bf16_t* s = vp + (size_t)(key0 + vkey) * ND + vd0;
      *(bf16x8*)&lV[vkey*72 + vd0]     = *(const bf16x8*)s;
      *(bf16x8*)&lV[vkey*72 + vd0 + 8] = *(const bf16x8*)(s + 8);
    }
#else
    if (w == 0) __builtin_amdgcn_s_wait_tensorcnt((short)0);
#endif
    __syncthreads();

    // ---- scores: 4 key-subtiles x 2 k-steps of WMMA ----
    floatx8 s4[4];
    for (int nt = 0; nt < 4; ++nt) {
      s4[nt] = zero8();
      for (int ks = 0; ks < 2; ++ks) {
        bf16x16 kf;
        for (int s = 0; s < 16; ++s)
          kf[s] = lK[(ks*32 + hl*8 + (s & 7) + (s >> 3)*16) * 72 + nt*16 + lr];
        s4[nt] = wmma_bf16(qf[ks], kf, s4[nt]);
      }
    }

    // ---- additive bias with padding mask (the big 268 MB stream) ----
    for (int nt = 0; nt < 4; ++nt)
      for (int r = 0; r < 8; ++r) {
        int row = qrow + hl*8 + r;
        int col = key0 + nt*16 + lr;
        float bv = bias[((size_t)h * NS + row) * NS + col];
        s4[nt][r] += mask[(size_t)b * NS + col] ? bv : -3.402823466e38f;
      }

    // ---- online softmax (row = 16 lanes of one half-wave) ----
    for (int r = 0; r < 8; ++r) {
      float mx = fmaxf(fmaxf(s4[0][r], s4[1][r]), fmaxf(s4[2][r], s4[3][r]));
      for (int off = 1; off < 16; off <<= 1)
        mx = fmaxf(mx, __shfl_xor(mx, off, 32));
      float mnew = fmaxf(mrun[r], mx);
      float scale = __expf(mrun[r] - mnew);
      lrun[r] *= scale;
      for (int dt = 0; dt < 4; ++dt) o[dt][r] *= scale;
      float rs = 0.f;
      for (int nt = 0; nt < 4; ++nt) {
        float pv = __expf(s4[nt][r] - mnew);
        s4[nt][r] = pv;
        rs += pv;
      }
      for (int off = 1; off < 16; off <<= 1)
        rs += __shfl_xor(rs, off, 32);
      lrun[r] += rs;
      mrun[r] = mnew;
    }

    // ---- probs (C layout) -> per-wave LDS -> reload as A fragments ----
    for (int nt = 0; nt < 4; ++nt)
      for (int r = 0; r < 8; ++r)
        lP[w][(hl*8 + r)*72 + nt*16 + lr] = (bf16_t)s4[nt][r];
    asm volatile("s_wait_dscnt 0x0" ::: "memory"); // intra-wave DS RAW

    // ---- P x V accumulate ----
    for (int ks = 0; ks < 2; ++ks) {
      const bf16_t* pa = &lP[w][lr*72 + ks*32 + hl*8];
      bf16x16 pf = frag2(pa, pa + 16);
      for (int dt = 0; dt < 4; ++dt) {
        bf16x16 vf;
        for (int s = 0; s < 16; ++s)
          vf[s] = lV[(ks*32 + hl*8 + (s & 7) + (s >> 3)*16) * 72 + dt*16 + lr];
        o[dt] = wmma_bf16(pf, vf, o[dt]);
      }
    }
  }

  // ---- normalize + store bf16 attention out in [b, s, h*D+d] ----
  for (int dt = 0; dt < 4; ++dt)
    for (int r = 0; r < 8; ++r) {
      int srow = qrow + hl*8 + r;
      int col  = h*ND + dt*16 + lr;
      float val = o[dt][r] / lrun[r];
      aout[(size_t)(b*NS + srow) * (NH*ND) + col] = (bf16_t)val;
    }
}

// ---------------------------------------------------------------------------
extern "C" void kernel_launch(void* const* d_in, const int* in_sizes, int n_in,
                              void* d_out, int out_size, void* d_ws, size_t ws_size,
                              hipStream_t stream)
{
  const float*         x     = (const float*)d_in[0];
  const float*         bias  = (const float*)d_in[1];
  const unsigned char* mask  = (const unsigned char*)d_in[2];
  const float*         w_qkv = (const float*)d_in[3];
  const float*         w_o   = (const float*)d_in[4];
  float*               out   = (float*)d_out;

  bf16_t* qkv_ws = (bf16_t*)d_ws;                          // 3*B*H*S*D bf16
  bf16_t* a_ws   = qkv_ws + (size_t)3 * NB * NH * NS * ND; // B*S*H*D bf16

  dim3 blk(256);
  // 1) fused QKV projection: [4096,1024] x [3072,1024]^T -> qkv bf16
  gemm_nt<0><<<dim3((3*NH*ND)/128, (NB*NS)/128), blk, 0, stream>>>(
      (const void*)x, w_qkv, (void*)qkv_ws, NB*NS, 3*NH*ND, NHID);
  // 2) flash attention over (b,h,q-tile)
  attn_kernel<<<dim3(NS/128, NB*NH), blk, 0, stream>>>(qkv_ws, bias, mask, a_ws);
  // 3) output projection: [4096,1024] x [1024,1024]^T -> f32 d_out
  gemm_nt<1><<<dim3(NHID/128, (NB*NS)/128), blk, 0, stream>>>(
      (const void*)a_ws, w_o, (void*)out, NB*NS, NHID, NH*ND);
}
